// KeyNet_11381663335019
// MI455X (gfx1250) — matrix operand order repaired and checked by reading, not compiled, not run on hardware
//
#include <hip/hip_runtime.h>
#include <cstddef>

// ---------------- problem constants (match reference) ----------------
#define Bc   8
#define Tc   512
#define Dc   512
#define Hc   8
#define Lc   6
#define SDc  64
#define ADc  16
#define KDc  64
#define MIDc 256
#define MAXTc 2048
#define Sc   1024            // 2*T
#define DHc  64              // D/H

typedef __attribute__((ext_vector_type(16))) __bf16 v16bf;
typedef __attribute__((ext_vector_type(8)))  __bf16 v8bf;
typedef __attribute__((ext_vector_type(4)))  __bf16 v4bf;
typedef __attribute__((ext_vector_type(8)))  float  v8f;

// ---------------- bf16 fragment loaders (16-byte vector loads) ----------------
__device__ __forceinline__ v16bf cat8(v8bf lo, v8bf hi) {
  return __builtin_shufflevector(lo, hi, 0, 1, 2, 3, 4, 5, 6, 7,
                                         8, 9, 10, 11, 12, 13, 14, 15);
}

// A-operand fragment (16x32): lane holds row (lane&15), g = lane>>4.
// Element j of v16bf maps to K = (j<8 ? j + 8g : (j-8) + 16 + 8g).
__device__ __forceinline__ v16bf load_frag_a_bf16(const __bf16* __restrict__ p, int g) {
  v8bf lo = *reinterpret_cast<const v8bf*>(p + g * 8);
  v8bf hi = *reinterpret_cast<const v8bf*>(p + g * 8 + 16);
  return cat8(lo, hi);
}

// B-operand fragment (32x16): lane holds column (lane&15); element j -> K = (j<8 ? j + 8g : (j-8) + 16 + 8g).
__device__ __forceinline__ v16bf load_frag_b_row_bf16(const __bf16* __restrict__ p, int g) {
  v8bf lo = *reinterpret_cast<const v8bf*>(p + g * 8);
  v8bf hi = *reinterpret_cast<const v8bf*>(p + g * 8 + 16);
  return cat8(lo, hi);
}

// Strided-column source (V matrix: element k at p[k*stride]).
__device__ __forceinline__ v16bf load_frag_b_col_bf16(const __bf16* __restrict__ p,
                                                      int stride, int g) {
  v16bf b;
#pragma unroll
  for (int j = 0; j < 16; ++j)
    b[j] = p[(size_t)(j + 8 * g + (j >= 8 ? 8 : 0)) * stride];
  return b;
}

__device__ __forceinline__ v8f wmma_bf16(v16bf a, v16bf b, v8f c) {
  return __builtin_amdgcn_wmma_f32_16x16x32_bf16(false, a, false, b, (short)0, c,
                                                 false, false);
}

// ---------------- generic GEMM: C = act(A @ W^T + bias) (+ f32 residual) ------
// A: [M,K] bf16 lda, W: [N,K] bf16 ldw, C: [M,N] OutT ldc. K%32==0, N%64==0.
// One wave computes a 64x64 C tile: 4 A-frags x 4 B-frags -> 16 WMMA / K-step.
// act: 0 none, 1 relu, 2 gelu(erf).
template <typename OutT>
__global__ void gemm_kernel(const __bf16* __restrict__ A, int lda,
                            const __bf16* __restrict__ W, int ldw,
                            const float* __restrict__ bias,
                            const float* __restrict__ residual,
                            OutT* __restrict__ C, int ldc,
                            int M, int K, int act) {
  const int lane = threadIdx.x & 31;
  const int g    = lane >> 4;
  const int idx  = lane & 15;
  const int m0   = blockIdx.x * 64;
  const int n0   = blockIdx.y * 64;

  const __bf16* ap[4];
#pragma unroll
  for (int t = 0; t < 4; ++t) {
    int r = m0 + t * 16 + idx;
    if (r >= M) r = M - 1;                       // clamp for ragged M
    ap[t] = A + (size_t)r * lda;
  }
  const __bf16* wp[4];
#pragma unroll
  for (int t = 0; t < 4; ++t)
    wp[t] = W + (size_t)(n0 + t * 16 + idx) * ldw;

  v8f acc[16];
#pragma unroll
  for (int i = 0; i < 16; ++i) acc[i] = (v8f){};

  for (int k0 = 0; k0 < K; k0 += 32) {
    __builtin_prefetch(ap[0] + k0 + 256, 0, 1);
    v16bf a0 = load_frag_a_bf16(ap[0] + k0, g);
    v16bf a1 = load_frag_a_bf16(ap[1] + k0, g);
    v16bf a2 = load_frag_a_bf16(ap[2] + k0, g);
    v16bf a3 = load_frag_a_bf16(ap[3] + k0, g);
#pragma unroll
    for (int nt = 0; nt < 4; ++nt) {
      v16bf bf = load_frag_b_row_bf16(wp[nt] + k0, g);
      acc[0 * 4 + nt] = wmma_bf16(a0, bf, acc[0 * 4 + nt]);
      acc[1 * 4 + nt] = wmma_bf16(a1, bf, acc[1 * 4 + nt]);
      acc[2 * 4 + nt] = wmma_bf16(a2, bf, acc[2 * 4 + nt]);
      acc[3 * 4 + nt] = wmma_bf16(a3, bf, acc[3 * 4 + nt]);
    }
  }

  float bb[4];
#pragma unroll
  for (int nt = 0; nt < 4; ++nt)
    bb[nt] = bias ? bias[n0 + nt * 16 + idx] : 0.0f;

#pragma unroll
  for (int mt = 0; mt < 4; ++mt) {
#pragma unroll
    for (int r = 0; r < 8; ++r) {
      const int row = m0 + mt * 16 + r + 8 * g;
      if (row >= M) continue;
      float v[4];
#pragma unroll
      for (int nt = 0; nt < 4; ++nt) v[nt] = acc[mt * 4 + nt][r] + bb[nt];
      if (act == 1) {
#pragma unroll
        for (int nt = 0; nt < 4; ++nt) v[nt] = fmaxf(v[nt], 0.0f);
      } else if (act == 2) {
#pragma unroll
        for (int nt = 0; nt < 4; ++nt)
          v[nt] = 0.5f * v[nt] * (1.0f + erff(v[nt] * 0.70710678118f));
      }
      if (residual) {
        const float* rp = residual + (size_t)row * ldc + n0 + idx;
#pragma unroll
        for (int nt = 0; nt < 4; ++nt) v[nt] += rp[nt * 16];
      }
      OutT* cp = C + (size_t)row * ldc + n0 + idx;
#pragma unroll
      for (int nt = 0; nt < 4; ++nt) cp[nt * 16] = (OutT)v[nt];
    }
  }
}

// ---------------- flash attention (causal, one wave / 16-row q tile) ---------
// q,k,v,y: [B,S,D] bf16, head h at column offset h*DH. grid = (S/16, H, B).
__global__ void attn_kernel(const __bf16* __restrict__ q,
                            const __bf16* __restrict__ k,
                            const __bf16* __restrict__ v,
                            __bf16* __restrict__ y) {
  __shared__ __bf16 pbuf[16 * 32];
  const int lane = threadIdx.x & 31;
  const int g    = lane >> 4;
  const int idx  = lane & 15;
  const int qt = blockIdx.x, h = blockIdx.y, b = blockIdx.z;
  const size_t base = ((size_t)b * Sc) * Dc + (size_t)h * DHc;

  const __bf16* qrow = q + base + (size_t)(qt * 16 + idx) * Dc;
  const v16bf qa0 = load_frag_a_bf16(qrow,      g);
  const v16bf qa1 = load_frag_a_bf16(qrow + 32, g);

  v8f o0 = {}, o1 = {}, o2 = {}, o3 = {};
  float mrow[8], srow[8];
#pragma unroll
  for (int r = 0; r < 8; ++r) { mrow[r] = -1e30f; srow[r] = 0.0f; }

  const int qmax  = qt * 16 + 15;
  const int ntile = qmax / 32 + 1;       // 32-key tiles (uniform per wave)

  for (int kt = 0; kt < ntile; ++kt) {
    v8f s0 = {}, s1 = {};
    {
      const __bf16* k0p = k + base + (size_t)(kt * 32 + idx) * Dc;
      s0 = wmma_bf16(qa0, load_frag_b_row_bf16(k0p,      g), s0);
      s0 = wmma_bf16(qa1, load_frag_b_row_bf16(k0p + 32, g), s0);
      const __bf16* k1p = k + base + (size_t)(kt * 32 + 16 + idx) * Dc;
      s1 = wmma_bf16(qa0, load_frag_b_row_bf16(k1p,      g), s1);
      s1 = wmma_bf16(qa1, load_frag_b_row_bf16(k1p + 32, g), s1);
    }

    float m_new[8];
#pragma unroll
    for (int r = 0; r < 8; ++r) {
      const int qi  = qt * 16 + r + 8 * g;
      const int kc0 = kt * 32 + idx;
      float a0 = s0[r] * 0.125f; if (kc0      > qi) a0 = -1e30f;
      float a1 = s1[r] * 0.125f; if (kc0 + 16 > qi) a1 = -1e30f;
      s0[r] = a0; s1[r] = a1;
      float mx = fmaxf(a0, a1);
#pragma unroll
      for (int off = 1; off < 16; off <<= 1) mx = fmaxf(mx, __shfl_xor(mx, off, 32));
      m_new[r] = fmaxf(mrow[r], mx);
    }

    __syncthreads();   // protect pbuf WAR across iterations (1 wave -> cheap)
#pragma unroll
    for (int r = 0; r < 8; ++r) {
      const float scl = __expf(mrow[r] - m_new[r]);
      const float p0 = __expf(s0[r] - m_new[r]);
      const float p1 = __expf(s1[r] - m_new[r]);
      float rs = p0 + p1;
#pragma unroll
      for (int off = 1; off < 16; off <<= 1) rs += __shfl_xor(rs, off, 32);
      srow[r] = srow[r] * scl + rs;
      mrow[r] = m_new[r];
      o0[r] *= scl; o1[r] *= scl; o2[r] *= scl; o3[r] *= scl;
      const int prow = r + 8 * g;
      pbuf[prow * 32 + idx]      = (__bf16)p0;
      pbuf[prow * 32 + 16 + idx] = (__bf16)p1;
    }
    __syncthreads();

    // reload P in A-fragment layout
    v16bf pa;
#pragma unroll
    for (int i = 0; i < 8; ++i) {
#pragma unroll
      for (int e = 0; e < 2; ++e) {
        int kk = ((i & 3) * 2 + e) + ((i >> 2) * 16) + g * 8;
        pa[2 * i + e] = pbuf[idx * 32 + kk];
      }
    }

    // O += P @ V   (V tile: 32 keys x 64 dh, column stride = D)
    const __bf16* vb = v + base + (size_t)(kt * 32) * Dc + idx;
    o0 = wmma_bf16(pa, load_frag_b_col_bf16(vb +  0, Dc, g), o0);
    o1 = wmma_bf16(pa, load_frag_b_col_bf16(vb + 16, Dc, g), o1);
    o2 = wmma_bf16(pa, load_frag_b_col_bf16(vb + 32, Dc, g), o2);
    o3 = wmma_bf16(pa, load_frag_b_col_bf16(vb + 48, Dc, g), o3);
  }

#pragma unroll
  for (int r = 0; r < 8; ++r) {
    const int row = qt * 16 + r + 8 * g;
    const float inv = 1.0f / srow[r];
    __bf16* yp = y + base + (size_t)row * Dc;
    yp[ 0 + idx] = (__bf16)(o0[r] * inv);
    yp[16 + idx] = (__bf16)(o1[r] * inv);
    yp[32 + idx] = (__bf16)(o2[r] * inv);
    yp[48 + idx] = (__bf16)(o3[r] * inv);
  }
}

// ---------------- layernorm: f32 in, bf16 out; one wave per 512-elem row -----
__global__ void ln_kernel(const float* __restrict__ x,
                          const float* __restrict__ w,
                          const float* __restrict__ bvec,
                          __bf16* __restrict__ out, int nrows) {
  const int lane = threadIdx.x & 31;
  const int wv   = threadIdx.x >> 5;
  const int row  = blockIdx.x * (blockDim.x >> 5) + wv;
  if (row >= nrows) return;
  const float* xp = x + (size_t)row * Dc;
  float vals[16];
  float s = 0.0f;
#pragma unroll
  for (int i = 0; i < 16; ++i) { vals[i] = xp[lane + 32 * i]; s += vals[i]; }
#pragma unroll
  for (int off = 1; off < 32; off <<= 1) s += __shfl_xor(s, off, 32);
  const float mean = s * (1.0f / Dc);
  float vs = 0.0f;
#pragma unroll
  for (int i = 0; i < 16; ++i) { float d = vals[i] - mean; vs += d * d; }
#pragma unroll
  for (int off = 1; off < 32; off <<= 1) vs += __shfl_xor(vs, off, 32);
  const float inv = rsqrtf(vs * (1.0f / Dc) + 1e-5f);
  __bf16* op = out + (size_t)row * Dc;
#pragma unroll
  for (int i = 0; i < 16; ++i) {
    int c = lane + 32 * i;
    op[c] = (__bf16)((vals[i] - mean) * inv * w[c] + bvec[c]);
  }
}

// ---------------- token assembly: interleave + positional (f32) --------------
__global__ void assemble_kernel(const float* __restrict__ se,
                                const float* __restrict__ ae,
                                const float* __restrict__ lp,
                                const float* __restrict__ gp,
                                const int* __restrict__ ts,
                                float* __restrict__ x) {
  size_t i = (size_t)blockIdx.x * blockDim.x + threadIdx.x;
  const size_t total = (size_t)Bc * Sc * Dc;
  if (i >= total) return;
  int d = (int)(i % Dc);
  size_t bs = i / Dc;
  int s = (int)(bs % Sc);
  int b = (int)(bs / Sc);
  int t = s >> 1;
  float tok = 0.0f;
  if ((s & 1) == 0)        tok = se[((size_t)b * Tc + t) * Dc + d];
  else if (t < Tc - 1)     tok = ae[((size_t)b * (Tc - 1) + t) * Dc + d];
  float l  = lp[(size_t)t * Dc + d];
  float gg = gp[(size_t)ts[b] * Dc + d];
  x[i] = tok + l + gg;
}

// ---------------- f32 -> bf16 conversion (n % 4 == 0) ----------------
__global__ void cvt_f32_bf16_kernel(const float* __restrict__ src,
                                    __bf16* __restrict__ dst, size_t n4) {
  size_t i = (size_t)blockIdx.x * blockDim.x + threadIdx.x;
  if (i >= n4) return;
  float4 f = reinterpret_cast<const float4*>(src)[i];
  v4bf o;
  o[0] = (__bf16)f.x; o[1] = (__bf16)f.y; o[2] = (__bf16)f.z; o[3] = (__bf16)f.w;
  reinterpret_cast<v4bf*>(dst)[i] = o;
}

// ---------------- zero-pad rows 16 -> 32 with f32->bf16 ----------------
__global__ void pad16to32_bf16_kernel(const float* __restrict__ src,
                                      __bf16* __restrict__ dst, int rows) {
  int i = blockIdx.x * blockDim.x + threadIdx.x;
  if (i >= rows * 32) return;
  int c = i & 31;
  int r = i >> 5;
  dst[i] = (c < ADc) ? (__bf16)src[r * ADc + c] : (__bf16)0.0f;
}

// ---------------- gathers for heads (bf16) ----------------
__global__ void gather_even_kernel(const __bf16* __restrict__ xin, __bf16* __restrict__ out) {
  size_t i = (size_t)blockIdx.x * blockDim.x + threadIdx.x;
  const size_t total = (size_t)Bc * Tc * Dc;
  if (i >= total) return;
  int d = (int)(i % Dc);
  size_t bt = i / Dc;
  int t = (int)(bt % Tc), b = (int)(bt / Tc);
  out[i] = xin[((size_t)b * Sc + 2 * t) * Dc + d];
}
__global__ void gather_odd_kernel(const __bf16* __restrict__ xin, __bf16* __restrict__ out) {
  size_t i = (size_t)blockIdx.x * blockDim.x + threadIdx.x;
  const size_t total = (size_t)Bc * (Tc - 1) * Dc;
  if (i >= total) return;
  int d = (int)(i % Dc);
  size_t bt = i / Dc;
  int t = (int)(bt % (Tc - 1)), b = (int)(bt / (Tc - 1));
  out[i] = xin[((size_t)b * Sc + 2 * t + 1) * Dc + d];
}

// ---------------- host side ----------------
static inline void launch_gemm_f(hipStream_t st, const __bf16* A, int lda,
                                 const __bf16* W, int ldw, const float* bias,
                                 const float* res, float* C, int ldc,
                                 int M, int N, int K, int act) {
  dim3 grid((M + 63) / 64, N / 64);
  gemm_kernel<float><<<grid, 32, 0, st>>>(A, lda, W, ldw, bias, res, C, ldc, M, K, act);
}
static inline void launch_gemm_b(hipStream_t st, const __bf16* A, int lda,
                                 const __bf16* W, int ldw, const float* bias,
                                 __bf16* C, int ldc, int M, int N, int K, int act) {
  dim3 grid((M + 63) / 64, N / 64);
  gemm_kernel<__bf16><<<grid, 32, 0, st>>>(A, lda, W, ldw, bias, nullptr, C, ldc, M, K, act);
}
static inline void launch_cvt(hipStream_t st, const float* src, __bf16* dst, size_t n) {
  size_t n4 = n / 4;
  cvt_f32_bf16_kernel<<<(unsigned)((n4 + 255) / 256), 256, 0, st>>>(src, dst, n4);
}

extern "C" void kernel_launch(void* const* d_in, const int* in_sizes, int n_in,
                              void* d_out, int out_size, void* d_ws, size_t ws_size,
                              hipStream_t stream) {
  (void)in_sizes; (void)n_in; (void)out_size; (void)ws_size;
  const float* states     = (const float*)d_in[0];
  const int*   timesteps  = (const int*)  d_in[1];
  const float* actions    = (const float*)d_in[2];
  const float* local_pos  = (const float*)d_in[3];
  const float* global_pos = (const float*)d_in[4];
  const float* se_w1 = (const float*)d_in[5];  const float* se_b1 = (const float*)d_in[6];
  const float* se_w2 = (const float*)d_in[7];  const float* se_b2 = (const float*)d_in[8];
  const float* ae_w1 = (const float*)d_in[9];  const float* ae_b1 = (const float*)d_in[10];
  const float* ae_w2 = (const float*)d_in[11]; const float* ae_b2 = (const float*)d_in[12];
  const float* ln1_w = (const float*)d_in[13]; const float* ln1_b = (const float*)d_in[14];
  const float* wq = (const float*)d_in[15]; const float* bq = (const float*)d_in[16];
  const float* wk = (const float*)d_in[17]; const float* bk = (const float*)d_in[18];
  const float* wv = (const float*)d_in[19]; const float* bv = (const float*)d_in[20];
  const float* wp = (const float*)d_in[21]; const float* bp = (const float*)d_in[22];
  const float* ln2_w = (const float*)d_in[23]; const float* ln2_b = (const float*)d_in[24];
  const float* w1 = (const float*)d_in[25]; const float* b1 = (const float*)d_in[26];
  const float* w2 = (const float*)d_in[27]; const float* b2 = (const float*)d_in[28];
  const float* lnf_w = (const float*)d_in[29]; const float* lnf_b = (const float*)d_in[30];
  const float* kp_w1 = (const float*)d_in[31]; const float* kp_b1 = (const float*)d_in[32];
  const float* kp_w2 = (const float*)d_in[33]; const float* kp_b2 = (const float*)d_in[34];
  const float* kp_w3 = (const float*)d_in[35]; const float* kp_b3 = (const float*)d_in[36];
  const float* sp_w1 = (const float*)d_in[37]; const float* sp_b1 = (const float*)d_in[38];
  const float* sp_w2 = (const float*)d_in[39]; const float* sp_b2 = (const float*)d_in[40];
  const float* sp_w3 = (const float*)d_in[41]; const float* sp_b3 = (const float*)d_in[42];

  const int BT  = Bc * Tc;            // 4096
  const int BT1 = Bc * (Tc - 1);      // 4088
  const int BS  = Bc * Sc;            // 8192
  const size_t BSD = (size_t)BS * Dc; // 4M elements

  // ---------------- workspace carving ----------------
  char* base = (char*)d_ws;
  float*  X   = (float*) base; base += BSD * 4;        // f32 residual stream
  __bf16* Hb  = (__bf16*)base; base += BSD * 2;        // LN out (bf16)
  __bf16* Qb  = (__bf16*)base; base += BSD * 2;
  __bf16* Kb  = (__bf16*)base; base += BSD * 2;
  __bf16* Vb  = (__bf16*)base; base += BSD * 2;
  __bf16* Yb  = (__bf16*)base; base += BSD * 2;        // attention out
  __bf16* MBb = (__bf16*)base; base += 4 * BSD * 2;    // mlp hidden / scratch
  __bf16* wcur = (__bf16*)base;                        // bf16 weights region

  auto carve = [&wcur](size_t n) { __bf16* p = wcur; wcur += (n + 7) & ~(size_t)7; return p; };
  __bf16* se_w1b = carve((size_t)MIDc * SDc);
  __bf16* se_w2b = carve((size_t)Dc * MIDc);
  __bf16* ae_w1p = carve((size_t)MIDc * 32);
  __bf16* ae_w2b = carve((size_t)Dc * MIDc);
  __bf16* wq_b   = carve((size_t)Lc * Dc * Dc);
  __bf16* wk_b   = carve((size_t)Lc * Dc * Dc);
  __bf16* wv_b   = carve((size_t)Lc * Dc * Dc);
  __bf16* wp_b   = carve((size_t)Lc * Dc * Dc);
  __bf16* w1_b   = carve((size_t)Lc * 4 * Dc * Dc);
  __bf16* w2_b   = carve((size_t)Lc * 4 * Dc * Dc);
  __bf16* kp_w1b = carve((size_t)MIDc * Dc);
  __bf16* kp_w2b = carve((size_t)MIDc * MIDc);
  __bf16* kp_w3b = carve((size_t)KDc * MIDc);
  __bf16* sp_w1b = carve((size_t)MIDc * Dc);
  __bf16* sp_w2b = carve((size_t)MIDc * MIDc);
  __bf16* sp_w3b = carve((size_t)SDc * MIDc);
  __bf16* states_b = carve((size_t)BT * SDc);
  __bf16* act_pad  = carve((size_t)BT1 * 32);

  // ---------------- one-time (per launch) bf16 staging ----------------
  launch_cvt(stream, se_w1, se_w1b, (size_t)MIDc * SDc);
  launch_cvt(stream, se_w2, se_w2b, (size_t)Dc * MIDc);
  launch_cvt(stream, ae_w2, ae_w2b, (size_t)Dc * MIDc);
  launch_cvt(stream, wq, wq_b, (size_t)Lc * Dc * Dc);
  launch_cvt(stream, wk, wk_b, (size_t)Lc * Dc * Dc);
  launch_cvt(stream, wv, wv_b, (size_t)Lc * Dc * Dc);
  launch_cvt(stream, wp, wp_b, (size_t)Lc * Dc * Dc);
  launch_cvt(stream, w1, w1_b, (size_t)Lc * 4 * Dc * Dc);
  launch_cvt(stream, w2, w2_b, (size_t)Lc * 4 * Dc * Dc);
  launch_cvt(stream, kp_w1, kp_w1b, (size_t)MIDc * Dc);
  launch_cvt(stream, kp_w2, kp_w2b, (size_t)MIDc * MIDc);
  launch_cvt(stream, kp_w3, kp_w3b, (size_t)KDc * MIDc);
  launch_cvt(stream, sp_w1, sp_w1b, (size_t)MIDc * Dc);
  launch_cvt(stream, sp_w2, sp_w2b, (size_t)MIDc * MIDc);
  launch_cvt(stream, sp_w3, sp_w3b, (size_t)SDc * MIDc);
  launch_cvt(stream, states, states_b, (size_t)BT * SDc);
  pad16to32_bf16_kernel<<<(BT1 * 32 + 255) / 256, 256, 0, stream>>>(actions, act_pad, BT1);
  pad16to32_bf16_kernel<<<(MIDc * 32 + 255) / 256, 256, 0, stream>>>(ae_w1, ae_w1p, MIDc);

  // ---------------- embeddings (MBb region as scratch) ----------------
  __bf16* se_hid = MBb;                                 // [BT, MID] bf16
  __bf16* ae_hid = MBb + (size_t)BT * MIDc;             // [BT1, MID] bf16
  float*  se_out = (float*)(MBb + 2 * (size_t)BT * MIDc); // [BT, D] f32
  float*  ae_out = se_out + (size_t)BT * Dc;            // [BT1, D] f32

  launch_gemm_b(stream, states_b, SDc, se_w1b, SDc, se_b1, se_hid, MIDc, BT, MIDc, SDc, 1);
  launch_gemm_f(stream, se_hid, MIDc, se_w2b, MIDc, se_b2, nullptr, se_out, Dc, BT, Dc, MIDc, 0);
  launch_gemm_b(stream, act_pad, 32, ae_w1p, 32, ae_b1, ae_hid, MIDc, BT1, MIDc, 32, 1);
  launch_gemm_f(stream, ae_hid, MIDc, ae_w2b, MIDc, ae_b2, nullptr, ae_out, Dc, BT1, Dc, MIDc, 0);

  {
    size_t total = (size_t)Bc * Sc * Dc;
    assemble_kernel<<<(unsigned)((total + 255) / 256), 256, 0, stream>>>(
        se_out, ae_out, local_pos, global_pos, timesteps, X);
  }

  // ---------------- transformer blocks ----------------
  for (int l = 0; l < Lc; ++l) {
    const size_t oD = (size_t)l * Dc, oDD = (size_t)l * Dc * Dc;
    const size_t oF = (size_t)l * 4 * Dc, oFD = (size_t)l * 4 * Dc * Dc;

    ln_kernel<<<BS / 8, 256, 0, stream>>>(X, ln1_w + oD, ln1_b + oD, Hb, BS);
    launch_gemm_b(stream, Hb, Dc, wq_b + oDD, Dc, bq + oD, Qb, Dc, BS, Dc, Dc, 0);
    launch_gemm_b(stream, Hb, Dc, wk_b + oDD, Dc, bk + oD, Kb, Dc, BS, Dc, Dc, 0);
    launch_gemm_b(stream, Hb, Dc, wv_b + oDD, Dc, bv + oD, Vb, Dc, BS, Dc, Dc, 0);

    attn_kernel<<<dim3(Sc / 16, Hc, Bc), 32, 0, stream>>>(Qb, Kb, Vb, Yb);

    // x = x + y @ Wp^T + bp   (f32 residual, in-place)
    launch_gemm_f(stream, Yb, Dc, wp_b + oDD, Dc, bp + oD, X, X, Dc, BS, Dc, Dc, 0);

    ln_kernel<<<BS / 8, 256, 0, stream>>>(X, ln2_w + oD, ln2_b + oD, Hb, BS);
    launch_gemm_b(stream, Hb, Dc, w1_b + oFD, Dc, b1 + oF, MBb, 4 * Dc, BS, 4 * Dc, Dc, 2);
    launch_gemm_f(stream, MBb, 4 * Dc, w2_b + oFD, 4 * Dc, b2 + oD, X, X, Dc, BS, Dc, 4 * Dc, 0);
  }

  // ---------------- final LN + heads ----------------
  ln_kernel<<<BS / 8, 256, 0, stream>>>(X, lnf_w, lnf_b, Hb, BS);

  __bf16* pe  = MBb;                                  // [BT, D] bf16
  __bf16* po  = MBb + (size_t)BT * Dc;                // [BT1, D] bf16
  __bf16* hh1 = MBb + 2 * (size_t)BT * Dc;            // [BT, MID] bf16
  __bf16* hh2 = hh1 + (size_t)BT * MIDc;              // [BT, MID] bf16

  {
    size_t te = (size_t)BT * Dc;
    gather_even_kernel<<<(unsigned)((te + 255) / 256), 256, 0, stream>>>(Hb, pe);
    size_t to = (size_t)BT1 * Dc;
    gather_odd_kernel<<<(unsigned)((to + 255) / 256), 256, 0, stream>>>(Hb, po);
  }

  float* out = (float*)d_out;
  // key head: [BT, D] -> MID -> MID -> KD
  launch_gemm_b(stream, pe, Dc, kp_w1b, Dc, kp_b1, hh1, MIDc, BT, MIDc, Dc, 1);
  launch_gemm_b(stream, hh1, MIDc, kp_w2b, MIDc, kp_b2, hh2, MIDc, BT, MIDc, MIDc, 1);
  launch_gemm_f(stream, hh2, MIDc, kp_w3b, MIDc, kp_b3, nullptr, out, KDc, BT, KDc, MIDc, 0);
  // state head: [BT1, D] -> MID -> MID -> SD
  launch_gemm_b(stream, po, Dc, sp_w1b, Dc, sp_b1, hh1, MIDc, BT1, MIDc, Dc, 1);
  launch_gemm_b(stream, hh1, MIDc, sp_w2b, MIDc, sp_b2, hh2, MIDc, BT1, MIDc, MIDc, 1);
  launch_gemm_f(stream, hh2, MIDc, sp_w3b, MIDc, sp_b3, nullptr, out + (size_t)BT * KDc, SDc,
              BT1, SDc, MIDc, 0);
}
